// Decoder_35029753266209
// MI455X (gfx1250) — matrix operand order repaired and checked by reading, not compiled, not run on hardware
//
#include <hip/hip_runtime.h>
#include <hip/hip_bf16.h>

typedef __attribute__((ext_vector_type(16))) __bf16 v16bf;
typedef __attribute__((ext_vector_type(8)))  __bf16 v8bf;
typedef __attribute__((ext_vector_type(8)))  float  v8f;

// ---------------------------------------------------------------------------
// WMMA bf16 GEMM, 4x M-register-blocked:
//   C[M,N] = A[M,K] (row-major) * Bt[N,K]^T + bias[N]
// One wave computes a 64-row x 16-col strip: per 32-wide K step it loads ONE
// B fragment and FOUR A fragments, issuing 4 v_wmma_f32_16x16x32_bf16 into
// 4 fp32 accumulators. B (the big weight) is read exactly once per GEMM.
// Fragment layouts (ISA 7.12.2):
//   A: lane m = lane&15, elems 0..7 = K(hi*8..+7), elems 8..15 = K(hi*8+16..+23)
//   B: lane n = lane&15, elems 0..7 = K(hi*8..+7), elems 8..15 = K(hi*8+16..+23)
//   C: VGPR r -> row +r+hi*8, col n0+(lane&15)
// Requires M % 64 == 0, N % 16 == 0, K % 32 == 0.
// ---------------------------------------------------------------------------
__global__ __launch_bounds__(256) void gemm_bf16_wmma(
    const __hip_bfloat16* __restrict__ A,
    const __hip_bfloat16* __restrict__ Bt,
    const float* __restrict__ bias,
    float* __restrict__ C,
    int M, int N, int K)
{
  const int wid  = (blockIdx.x * blockDim.x + threadIdx.x) >> 5;
  const int lane = threadIdx.x & 31;
  const int tM = M >> 6;                   // 64-row strips
  const int tN = N >> 4;                   // 16-col tiles
  if (wid >= tM * tN) return;              // wave-uniform: EXEC stays all-1s
  const int mt = wid % tM, nt = wid / tM;
  const int m0 = mt << 6, n0 = nt << 4;

  const int lo = lane & 15;
  const int hi = lane >> 4;

  const __bf16* Arow = (const __bf16*)(const void*)A  + (size_t)(m0 + lo) * K + hi * 8;
  const __bf16* Brow = (const __bf16*)(const void*)Bt + (size_t)(n0 + lo) * K + hi * 8;
  const size_t strip = (size_t)16 * K;     // 16 rows of A

  v8f acc0 = {}, acc1 = {}, acc2 = {}, acc3 = {};
  for (int k0 = 0; k0 < K; k0 += 32) {
    __builtin_prefetch((const void*)(Brow + k0 + 32), 0, 1);
    // one B fragment, reused 4x
    v8bf b_lo = *(const v8bf*)(Brow + k0);
    v8bf b_hi = *(const v8bf*)(Brow + k0 + 16);
    v16bf b;
#pragma unroll
    for (int i = 0; i < 8; ++i) { b[i] = b_lo[i]; b[i + 8] = b_hi[i]; }

    v16bf a0, a1, a2, a3;
    {
      v8bf l = *(const v8bf*)(Arow + k0),          h = *(const v8bf*)(Arow + k0 + 16);
#pragma unroll
      for (int i = 0; i < 8; ++i) { a0[i] = l[i]; a0[i + 8] = h[i]; }
    }
    {
      v8bf l = *(const v8bf*)(Arow + strip + k0),  h = *(const v8bf*)(Arow + strip + k0 + 16);
#pragma unroll
      for (int i = 0; i < 8; ++i) { a1[i] = l[i]; a1[i + 8] = h[i]; }
    }
    {
      v8bf l = *(const v8bf*)(Arow + 2 * strip + k0), h = *(const v8bf*)(Arow + 2 * strip + k0 + 16);
#pragma unroll
      for (int i = 0; i < 8; ++i) { a2[i] = l[i]; a2[i + 8] = h[i]; }
    }
    {
      v8bf l = *(const v8bf*)(Arow + 3 * strip + k0), h = *(const v8bf*)(Arow + 3 * strip + k0 + 16);
#pragma unroll
      for (int i = 0; i < 8; ++i) { a3[i] = l[i]; a3[i + 8] = h[i]; }
    }
    acc0 = __builtin_amdgcn_wmma_f32_16x16x32_bf16(false, a0, false, b, (short)0, acc0, false, false);
    acc1 = __builtin_amdgcn_wmma_f32_16x16x32_bf16(false, a1, false, b, (short)0, acc1, false, false);
    acc2 = __builtin_amdgcn_wmma_f32_16x16x32_bf16(false, a2, false, b, (short)0, acc2, false, false);
    acc3 = __builtin_amdgcn_wmma_f32_16x16x32_bf16(false, a3, false, b, (short)0, acc3, false, false);
  }

  const int col = n0 + lo;
  const float bv = bias ? bias[col] : 0.0f;
  float* Cb = C + (size_t)(m0 + hi * 8) * N + col;
#pragma unroll
  for (int r = 0; r < 8; ++r) Cb[(size_t)r * N] = acc0[r] + bv;
  Cb += (size_t)16 * N;
#pragma unroll
  for (int r = 0; r < 8; ++r) Cb[(size_t)r * N] = acc1[r] + bv;
  Cb += (size_t)16 * N;
#pragma unroll
  for (int r = 0; r < 8; ++r) Cb[(size_t)r * N] = acc2[r] + bv;
  Cb += (size_t)16 * N;
#pragma unroll
  for (int r = 0; r < 8; ++r) Cb[(size_t)r * N] = acc3[r] + bv;
}

// ---------------------------------------------------------------------------
// fp32 -> bf16 convert (same layout)
// ---------------------------------------------------------------------------
__global__ void k_cvt(const float* __restrict__ src, __hip_bfloat16* __restrict__ dst, int n)
{
  int i = blockIdx.x * blockDim.x + threadIdx.x;
  if (i < n) dst[i] = __float2bfloat16(src[i]);
}

// fp32 (rows x cols) -> bf16 transposed (cols x rows)
__global__ void k_cvt_t(const float* __restrict__ src, __hip_bfloat16* __restrict__ dst,
                        int rows, int cols)
{
  int i = blockIdx.x * blockDim.x + threadIdx.x;
  if (i >= rows * cols) return;
  int r = i / cols, c = i % cols;
  dst[(size_t)c * rows + r] = __float2bfloat16(src[i]);
}

// ---------------------------------------------------------------------------
// init: tok = SOS, h = encoder_final_state (fp32 + bf16 copies)
// ---------------------------------------------------------------------------
__global__ void k_init(int* tok, const float* __restrict__ efs,
                       float* __restrict__ h, __hip_bfloat16* __restrict__ h_bf,
                       int Bn, int nh)
{
  int i = blockIdx.x * blockDim.x + threadIdx.x;
  if (i < Bn) tok[i] = 1;  // SOS
  if (i < nh) { float v = efs[i]; h[i] = v; h_bf[i] = __float2bfloat16(v); }
}

// ---------------------------------------------------------------------------
// scores[s,b] = sum_a tanh(qW[b,a] + keys_proj[s,b,a]) * v_att[a]
// one block per (s,b)
// ---------------------------------------------------------------------------
__global__ __launch_bounds__(256) void k_scores(
    const float* __restrict__ qW, const float* __restrict__ keys_proj,
    const float* __restrict__ v_att, float* __restrict__ scores, int Bn, int A)
{
  int sb = blockIdx.x;
  int b = sb % Bn;
  int t = threadIdx.x;
  float part = 0.0f;
  for (int a = t; a < A; a += 256)
    part += tanhf(qW[(size_t)b * A + a] + keys_proj[(size_t)sb * A + a]) * v_att[a];
  __shared__ float red[256];
  red[t] = part; __syncthreads();
  for (int s = 128; s > 0; s >>= 1) { if (t < s) red[t] += red[t + s]; __syncthreads(); }
  if (t == 0) scores[sb] = red[0];
}

// ---------------------------------------------------------------------------
// softmax over S then ctx[b,k] = sum_s attn[s] * enc[s,b,k]; one block per b
// ---------------------------------------------------------------------------
__global__ __launch_bounds__(256) void k_attn_ctx(
    const float* __restrict__ scores, const float* __restrict__ enc,
    float* __restrict__ ctx, int S, int Bn, int Kd)
{
  int b = blockIdx.x;
  int t = threadIdx.x;
  __shared__ float red[256];
  __shared__ float attn[64];
  float v = (t < S) ? scores[(size_t)t * Bn + b] : -3.402823466e38f;
  red[t] = v; __syncthreads();
  for (int s = 128; s > 0; s >>= 1) { if (t < s) red[t] = fmaxf(red[t], red[t + s]); __syncthreads(); }
  float mx = red[0]; __syncthreads();
  float e = (t < S) ? __expf(v - mx) : 0.0f;
  red[t] = e; __syncthreads();
  for (int s = 128; s > 0; s >>= 1) { if (t < s) red[t] += red[t + s]; __syncthreads(); }
  float inv = 1.0f / red[0];
  if (t < S) attn[t] = e * inv;
  __syncthreads();
  for (int k = t; k < Kd; k += 256) {
    float acc = 0.0f;
    for (int s = 0; s < S; ++s)
      acc += attn[s] * enc[((size_t)s * Bn + b) * Kd + k];
    ctx[(size_t)b * Kd + k] = acc;
  }
}

// ---------------------------------------------------------------------------
// build rnn_in = [emb[tok], ctx] (bf16) ; also stash x_emb (f32) and the
// x_emb / ctx slices of mlp_in = [x_emb, h_new, ctx] (bf16)
// ---------------------------------------------------------------------------
__global__ void k_build_in(const int* __restrict__ tok, const float* __restrict__ emb,
                           const float* __restrict__ ctx,
                           float* __restrict__ x_emb,
                           __hip_bfloat16* __restrict__ rnn_in,
                           __hip_bfloat16* __restrict__ mlp_in,
                           int Bn, int E, int Kd, int H)
{
  int per = E + Kd;
  int i = blockIdx.x * blockDim.x + threadIdx.x;
  if (i >= Bn * per) return;
  int b = i / per, j = i % per;
  size_t mrow = (size_t)b * (E + H + Kd);
  if (j < E) {
    float v = emb[(size_t)tok[b] * E + j];
    x_emb[(size_t)b * E + j] = v;
    rnn_in[(size_t)b * per + j] = __float2bfloat16(v);
    mlp_in[mrow + j] = __float2bfloat16(v);
  } else {
    int k = j - E;
    float v = ctx[(size_t)b * Kd + k];
    rnn_in[(size_t)b * per + j] = __float2bfloat16(v);
    mlp_in[mrow + E + H + k] = __float2bfloat16(v);
  }
}

// ---------------------------------------------------------------------------
// GRU elementwise combine; updates h (f32 + bf16) and h slice of mlp_in
// ---------------------------------------------------------------------------
__global__ void k_gru(const float* __restrict__ gi, const float* __restrict__ gh,
                      float* __restrict__ h, __hip_bfloat16* __restrict__ h_bf,
                      __hip_bfloat16* __restrict__ mlp_in,
                      int Bn, int H, int mlp_ld, int E)
{
  int i = blockIdx.x * blockDim.x + threadIdx.x;
  if (i >= Bn * H) return;
  int b = i / H, j = i % H;
  const float* gib = gi + (size_t)b * 3 * H;
  const float* ghb = gh + (size_t)b * 3 * H;
  float r = 1.0f / (1.0f + __expf(-(gib[j] + ghb[j])));
  float z = 1.0f / (1.0f + __expf(-(gib[H + j] + ghb[H + j])));
  float n = tanhf(gib[2 * H + j] + r * ghb[2 * H + j]);
  float hn = (1.0f - z) * n + z * h[i];
  h[i] = hn;
  h_bf[i] = __float2bfloat16(hn);
  mlp_in[(size_t)b * mlp_ld + E + j] = __float2bfloat16(hn);
}

__global__ void k_tanh_bf(const float* __restrict__ in, __hip_bfloat16* __restrict__ out, int n)
{
  int i = blockIdx.x * blockDim.x + threadIdx.x;
  if (i < n) out[i] = __float2bfloat16(tanhf(in[i]));
}

// ---------------------------------------------------------------------------
// softmax over V + argmax -> probs written ONCE into d_out[b][t][:]
// (logits stay L2-resident; recompute exp instead of re-reading d_out)
// ---------------------------------------------------------------------------
__global__ __launch_bounds__(1024) void k_softmax_argmax(
    const float* __restrict__ logits, float* __restrict__ out, int* __restrict__ tok,
    int V, int T, int t_step)
{
  int b = blockIdx.x;
  const float* row = logits + (size_t)b * V;
  float* orow = out + ((size_t)b * T + t_step) * V;
  int t = threadIdx.x;
  __shared__ float smx[1024];
  __shared__ int   smi[1024];
  // pass 1: max + argmax
  float mx = -3.402823466e38f; int mi = 0;
  for (int v = t; v < V; v += 1024) {
    float x = row[v];
    if (x > mx) { mx = x; mi = v; }
  }
  smx[t] = mx; smi[t] = mi; __syncthreads();
  for (int s = 512; s > 0; s >>= 1) {
    if (t < s) {
      if (smx[t + s] > smx[t] || (smx[t + s] == smx[t] && smi[t + s] < smi[t])) {
        smx[t] = smx[t + s]; smi[t] = smi[t + s];
      }
    }
    __syncthreads();
  }
  float gmx = smx[0]; int gmi = smi[0];
  __syncthreads();
  // pass 2: sum of exp
  float part = 0.0f;
  for (int v = t; v < V; v += 1024) part += __expf(row[v] - gmx);
  smx[t] = part; __syncthreads();
  for (int s = 512; s > 0; s >>= 1) { if (t < s) smx[t] += smx[t + s]; __syncthreads(); }
  float inv = 1.0f / smx[0];
  // pass 3: single write of probs
  for (int v = t; v < V; v += 1024) orow[v] = __expf(row[v] - gmx) * inv;
  if (t == 0) tok[b] = gmi;
}

// ---------------------------------------------------------------------------
// host launcher
// ---------------------------------------------------------------------------
static inline void gemm_launch(const __hip_bfloat16* A, const __hip_bfloat16* Bt,
                               const float* bias, float* C, int M, int N, int K,
                               hipStream_t s)
{
  int waves = (M >> 6) * (N >> 4);     // 64-row strips x 16-col tiles
  int blocks = (waves + 7) / 8;
  gemm_bf16_wmma<<<blocks, 256, 0, s>>>(A, Bt, bias, C, M, N, K);
}

extern "C" void kernel_launch(void* const* d_in, const int* in_sizes, int n_in,
                              void* d_out, int out_size, void* d_ws, size_t ws_size,
                              hipStream_t stream) {
  const int S = 64, B = 64, Kd = 512, H = 512, E = 512, A = 512, V = 32000, T = 32;

  const float* enc   = (const float*)d_in[0];   // (S,B,K)
  const float* efs   = (const float*)d_in[1];   // (1,B,H)
  const float* emb   = (const float*)d_in[2];   // (V,E)
  const float* Wq    = (const float*)d_in[3];   // (H,A)
  const float* Wk    = (const float*)d_in[4];   // (K,A)
  const float* v_att = (const float*)d_in[5];   // (A,)
  const float* W_ih  = (const float*)d_in[6];   // (3H, E+K)
  const float* W_hh  = (const float*)d_in[7];   // (3H, H)
  const float* b_ih  = (const float*)d_in[8];
  const float* b_hh  = (const float*)d_in[9];
  const float* W1    = (const float*)d_in[10];  // (E+H+K, H)
  const float* b1    = (const float*)d_in[11];
  const float* W2    = (const float*)d_in[12];  // (H, V)
  const float* b2    = (const float*)d_in[13];
  float* out = (float*)d_out;

  // workspace arena (256B aligned slices)
  char* ws = (char*)d_ws;
  size_t off = 0;
  auto carve = [&](size_t bytes) -> void* {
    void* p = ws + off;
    off = (off + bytes + 255) & ~(size_t)255;
    return p;
  };
  __hip_bfloat16* enc_bf  = (__hip_bfloat16*)carve((size_t)S * B * Kd * 2);
  __hip_bfloat16* WkT     = (__hip_bfloat16*)carve((size_t)A * Kd * 2);
  __hip_bfloat16* WqT     = (__hip_bfloat16*)carve((size_t)A * H * 2);
  __hip_bfloat16* Wih_bf  = (__hip_bfloat16*)carve((size_t)3 * H * (E + Kd) * 2);
  __hip_bfloat16* Whh_bf  = (__hip_bfloat16*)carve((size_t)3 * H * H * 2);
  __hip_bfloat16* W1T     = (__hip_bfloat16*)carve((size_t)H * (E + H + Kd) * 2);
  __hip_bfloat16* W2T     = (__hip_bfloat16*)carve((size_t)V * H * 2);
  float* keys_proj = (float*)carve((size_t)S * B * A * 4);
  float* h_f32   = (float*)carve((size_t)B * H * 4);
  __hip_bfloat16* h_bf = (__hip_bfloat16*)carve((size_t)B * H * 2);
  float* qW      = (float*)carve((size_t)B * A * 4);
  float* scores  = (float*)carve((size_t)S * B * 4);
  float* ctx     = (float*)carve((size_t)B * Kd * 4);
  float* x_emb   = (float*)carve((size_t)B * E * 4);
  __hip_bfloat16* rnn_in = (__hip_bfloat16*)carve((size_t)B * (E + Kd) * 2);
  __hip_bfloat16* mlp_in = (__hip_bfloat16*)carve((size_t)B * (E + H + Kd) * 2);
  float* gi      = (float*)carve((size_t)B * 3 * H * 4);
  float* gh      = (float*)carve((size_t)B * 3 * H * 4);
  float* hid     = (float*)carve((size_t)B * H * 4);
  __hip_bfloat16* hid_bf = (__hip_bfloat16*)carve((size_t)B * H * 2);
  float* logits  = (float*)carve((size_t)B * V * 4);
  int* tok       = (int*)carve((size_t)B * 4);
  (void)ws_size; (void)in_sizes; (void)n_in; (void)out_size;

  auto cdiv = [](int a, int b) { return (a + b - 1) / b; };

  // one-time conversions (bf16; "B" operands stored as N x K row-major)
  k_cvt  <<<cdiv(S * B * Kd, 256), 256, 0, stream>>>(enc, enc_bf, S * B * Kd);
  k_cvt_t<<<cdiv(Kd * A, 256),    256, 0, stream>>>(Wk, WkT, Kd, A);          // (K,A)->(A,K)
  k_cvt_t<<<cdiv(H * A, 256),     256, 0, stream>>>(Wq, WqT, H, A);           // (H,A)->(A,H)
  k_cvt  <<<cdiv(3 * H * (E + Kd), 256), 256, 0, stream>>>(W_ih, Wih_bf, 3 * H * (E + Kd));
  k_cvt  <<<cdiv(3 * H * H, 256), 256, 0, stream>>>(W_hh, Whh_bf, 3 * H * H);
  k_cvt_t<<<cdiv((E + H + Kd) * H, 256), 256, 0, stream>>>(W1, W1T, E + H + Kd, H);
  k_cvt_t<<<cdiv(H * V, 256),     256, 0, stream>>>(W2, W2T, H, V);           // (H,V)->(V,H)

  // keys_proj[s,b,:] = enc[s,b,:] @ Wk   -> (S*B, A)
  gemm_launch(enc_bf, WkT, nullptr, keys_proj, S * B, A, Kd, stream);

  // init tok + h
  k_init<<<cdiv(B * H, 256), 256, 0, stream>>>(tok, efs, h_f32, h_bf, B, B * H);

  for (int t = 0; t < T; ++t) {
    // q @ Wq
    gemm_launch(h_bf, WqT, nullptr, qW, B, A, H, stream);
    // attention scores, softmax + context
    k_scores<<<S * B, 256, 0, stream>>>(qW, keys_proj, v_att, scores, B, A);
    k_attn_ctx<<<B, 256, 0, stream>>>(scores, enc, ctx, S, B, Kd);
    // rnn_in / mlp_in assembly (embedding gather uses tok from prev step)
    k_build_in<<<cdiv(B * (E + Kd), 256), 256, 0, stream>>>(
        tok, emb, ctx, x_emb, rnn_in, mlp_in, B, E, Kd, H);
    // GRU gates
    gemm_launch(rnn_in, Wih_bf, b_ih, gi, B, 3 * H, E + Kd, stream);
    gemm_launch(h_bf,   Whh_bf, b_hh, gh, B, 3 * H, H, stream);
    k_gru<<<cdiv(B * H, 256), 256, 0, stream>>>(gi, gh, h_f32, h_bf, mlp_in,
                                                B, H, E + H + Kd, E);
    // MLP
    gemm_launch(mlp_in, W1T, b1, hid, B, H, E + H + Kd, stream);
    k_tanh_bf<<<cdiv(B * H, 256), 256, 0, stream>>>(hid, hid_bf, B * H);
    // vocab projection (dominant GEMM, 64 x 32000 x 512)
    gemm_launch(hid_bf, W2T, b2, logits, B, V, H, stream);
    // softmax -> probs into d_out[b][t][:], argmax -> next tok
    k_softmax_argmax<<<B, 1024, 0, stream>>>(logits, out, tok, V, T, t);
  }
}
